// RetroFittedGPT2_79869211836497
// MI455X (gfx1250) — compile-verified, matrix operand order checked by reading, not run on hardware
//
#include <hip/hip_runtime.h>
#include <hip/hip_bf16.h>
#include <cstdint>
#include <cstddef>

// ---------------------------------------------------------------------------
// RETRO-fitted GPT-2 forward for MI455X (gfx1250, wave32, WMMA).
//
// Roofline: readout GEMM (2048x768x50257 ~ 158 GFLOP) dominates -> all GEMMs
// run on v_wmma_f32_16x16x32_f16 (f16 in, f32 accum). Weights are converted
// AND transposed to [N x K] f16 once per call so both A and B tiles stage
// into LDS with branchless contiguous b128 transfers. Tile staging uses
// gfx1250 global_load_async_to_lds_b128 (ASYNCcnt) with DOUBLE BUFFERING:
// the next K-step's tiles are in flight while the current step's WMMAs run,
// and there is a single s_barrier per K-step. Attention (~8 GFLOP total)
// stays on the VALU.
// ---------------------------------------------------------------------------

typedef __attribute__((ext_vector_type(16))) _Float16 v16h;
typedef __attribute__((ext_vector_type(8)))  _Float16 v8h;
typedef __attribute__((ext_vector_type(8)))  float    v8f;

#define NH   12
#define DK_  64
#define DMODEL 768
#define FFN_ 3072
#define NB   2
#define SEQ  1024
#define VOCAB 50257
#define CHL  64      // CL
#define NCH  16      // C
#define NNB  2       // N (neighbors)
#define NL_  128     // L

#define GF_RELU 1
#define GF_RES  2

#define USE_ASYNC_STAGE 1   // gfx1250 async global->LDS staging in the GEMM

// ---------------------------------------------------------------------------
// f32 -> f16 convert (activations)
// ---------------------------------------------------------------------------
__global__ void k_cvt_f16(const float* __restrict__ s, _Float16* __restrict__ d,
                          long long n) {
  long long i = (long long)blockIdx.x * 256 + threadIdx.x;
  if (i < n) d[i] = (_Float16)s[i];
}

// ---------------------------------------------------------------------------
// Weight convert + transpose: src [K x N] f32 -> dst [Npad x K] f16,
// batched over blockIdx.z (layer). Padded rows (n >= N) are zero-filled so
// the GEMM can load B tiles with no bounds checks.
// ---------------------------------------------------------------------------
__global__ __launch_bounds__(256)
void k_cvt_tr(const float* __restrict__ src, _Float16* __restrict__ dst,
              int K, int N, int Npad) {
  __shared__ float tile[32][33];
  const float* s = src + (size_t)blockIdx.z * K * N;
  _Float16* d = dst + (size_t)blockIdx.z * Npad * K;
  int bk = blockIdx.x * 32, bn = blockIdx.y * 32;
  int tx = threadIdx.x & 31, ty = threadIdx.x >> 5;   // 32 x 8
  #pragma unroll
  for (int r = ty; r < 32; r += 8) {
    int kk = bk + r, nn = bn + tx;
    tile[r][tx] = (kk < K && nn < N) ? s[(size_t)kk * N + nn] : 0.0f;
  }
  __syncthreads();
  #pragma unroll
  for (int r = ty; r < 32; r += 8) {
    int nn = bn + r, kk = bk + tx;
    if (kk < K && nn < Npad) d[(size_t)nn * K + kk] = (_Float16)tile[tx][r];
  }
}

// ---------------------------------------------------------------------------
// RMSNorm (optionally shifted/padded view for CCA): rows laid out as
// [batch, perBatch]; row i<valid reads src row (b*srcPerBatch + i + shift),
// rows i>=valid emit zeros. Output is f16 (feeds WMMA GEMMs).
// ---------------------------------------------------------------------------
__global__ __launch_bounds__(256)
void k_rmsnorm(const float* __restrict__ x, const float* __restrict__ g,
               _Float16* __restrict__ out, int perBatch, int valid, int shift,
               int srcPerBatch) {
  __shared__ float red[256];
  int r = blockIdx.x;
  int b = r / perBatch, i = r % perBatch;
  int tid = threadIdx.x;
  _Float16* orow = out + (long long)r * DMODEL;
  if (i >= valid) {
    for (int j = tid; j < DMODEL; j += 256) orow[j] = (_Float16)0.0f;
    return;
  }
  const float* xr = x + ((long long)b * srcPerBatch + i + shift) * DMODEL;
  float s = 0.0f;
  for (int j = tid; j < DMODEL; j += 256) { float v = xr[j]; s += v * v; }
  red[tid] = s; __syncthreads();
  for (int st = 128; st > 0; st >>= 1) {
    if (tid < st) red[tid] += red[tid + st];
    __syncthreads();
  }
  float scale = rsqrtf(red[0] / (float)DMODEL + 1e-8f);
  for (int j = tid; j < DMODEL; j += 256)
    orow[j] = (_Float16)(xr[j] * scale * g[j]);
}

// ---------------------------------------------------------------------------
// RoPE applied in place to the q and k thirds of a [rows, 3*NH*DK] buffer.
// One thread handles one rotation pair (d, d+32). Position = row % seqLen.
// ---------------------------------------------------------------------------
__global__ void k_rope(float* __restrict__ buf, long long rows, int seqLen) {
  long long t = (long long)blockIdx.x * 256 + threadIdx.x;
  long long total = rows * (long long)(NH * 32 * 2);
  if (t >= total) return;
  int d = (int)(t % 32); long long r = t / 32;
  int hh = (int)(r % NH); r /= NH;
  int part = (int)(r % 2); r /= 2;          // 0 = q, 1 = k
  long long row = r;
  int s = (int)(row % seqLen);
  float* base = buf + row * (3LL * NH * DK_) + part * (NH * DK_) + hh * DK_;
  float theta = powf(10000.0f, -(float)d / 32.0f);   // 10000^(-(2d)/64)
  float ang = (float)s * theta;
  float c = cosf(ang), sn = sinf(ang);
  float x0 = base[d], x1 = base[d + 32];
  base[d]      = x0 * c - x1 * sn;
  base[d + 32] = x1 * c + x0 * sn;
}

// ---------------------------------------------------------------------------
// Generic softmax attention (VALU): block = one (q-batch, head, query row).
//   kv batch index = q batch index / kvDiv.
// ---------------------------------------------------------------------------
__global__ __launch_bounds__(256)
void k_attn(const float* __restrict__ q, const float* __restrict__ k,
            const float* __restrict__ v, float* __restrict__ o,
            int Lq, int Lkv, int kvDiv, int causal, float scale,
            long long qRow, long long qBatch, long long kRow, long long kBatch,
            long long vRow, long long vBatch, long long oRow, long long oBatch) {
  __shared__ float sc[1024];
  __shared__ float qsh[DK_];
  __shared__ float red[256];
  int tid = threadIdx.x;
  int g = blockIdx.x;                  // qb*NH + h
  int qi = blockIdx.y;
  int qb = g / NH, h = g % NH;
  int kb = qb / kvDiv;
  const float* qrow = q + (long long)qb * qBatch + (long long)qi * qRow + h * DK_;
  const float* kbas = k + (long long)kb * kBatch + h * DK_;
  const float* vbas = v + (long long)kb * vBatch + h * DK_;
  float* orow = o + (long long)qb * oBatch + (long long)qi * oRow + h * DK_;

  if (tid < DK_) qsh[tid] = qrow[tid];
  __syncthreads();

  int jmax = causal ? (qi + 1) : Lkv;
  float lm = -1e30f;
  for (int j = tid; j < Lkv; j += 256) {
    float sv = -1e30f;
    if (j < jmax) {
      const float* kr = kbas + (long long)j * kRow;
      float dot = 0.0f;
      #pragma unroll
      for (int d2 = 0; d2 < DK_; ++d2) dot += kr[d2] * qsh[d2];
      sv = dot * scale;
    }
    sc[j] = sv;
    lm = fmaxf(lm, sv);
  }
  red[tid] = lm; __syncthreads();
  for (int st = 128; st > 0; st >>= 1) {
    if (tid < st) red[tid] = fmaxf(red[tid], red[tid + st]);
    __syncthreads();
  }
  float m = red[0]; __syncthreads();

  float ls = 0.0f;
  for (int j = tid; j < Lkv; j += 256) {
    float p = __expf(sc[j] - m);
    sc[j] = p; ls += p;
  }
  red[tid] = ls; __syncthreads();
  for (int st = 128; st > 0; st >>= 1) {
    if (tid < st) red[tid] += red[tid + st];
    __syncthreads();
  }
  float inv = 1.0f / red[0]; __syncthreads();

  int d = tid & 63, part = tid >> 6;
  float acc = 0.0f;
  for (int j = part; j < Lkv; j += 4)
    acc += sc[j] * vbas[(long long)j * vRow + d];
  red[tid] = acc; __syncthreads();
  if (tid < DK_)
    orow[tid] = (red[tid] + red[tid + 64] + red[tid + 128] + red[tid + 192]) * inv;
}

// ---------------------------------------------------------------------------
// WMMA GEMM: C[M,N] = A[M,K](f16,row-major) * Bt[Npad,K](f16, pre-transposed)
//            (+bias)(+relu)(+residual)
// Host contract: M % 64 == 0, K % 32 == 0, Bt padded to Npad = ceil(N/128)*128
// rows (pad rows zero). Only the C store checks col<N.
// Block tile 64x128, 8 waves, each wave a 16x64 slab (4 wmma / K-step).
// LDS rows padded to 40 halfs (80B stride) -> conflict-free b128 access; both
// fragments load as two contiguous b128 DS reads matching the 16-bit 16x32
// lane layout: lane l(0..15): K {0..7,16..23}; lane l+16: K {8..15,24..31}.
// Double-buffered async staging: step k+1's global->LDS transfers run behind
// step k's WMMAs; one s_barrier per K-step. Writes to buffer b are issued
// only after the barrier that follows the last reads of b (each wave's
// fragment ds_loads complete before it signals: the wmma forces dscnt wait).
// ---------------------------------------------------------------------------
#define GBM 64
#define GBN 128
#define GBK 32
#define LSH 40

__device__ __forceinline__ void stage_b128(const _Float16* gsrc, _Float16* lds) {
#if USE_ASYNC_STAGE
  unsigned ldsa = (unsigned)(uintptr_t)lds;                 // low 32b = LDS offset
  unsigned long long ga = (unsigned long long)(uintptr_t)gsrc;
  asm volatile("global_load_async_to_lds_b128 %0, %1, off"
               :: "v"(ldsa), "v"(ga) : "memory");
#else
  *(v8h*)lds = *(const v8h*)gsrc;
#endif
}

__global__ __launch_bounds__(256)
void k_gemm_f16(const _Float16* __restrict__ A, const _Float16* __restrict__ Bt,
                const float* __restrict__ bias, const float* __restrict__ resid,
                float* __restrict__ C, int M, int N, int K, int flags) {
  __shared__ _Float16 As[2][GBM * LSH];
  __shared__ _Float16 Bs[2][GBN * LSH];
  int tid = threadIdx.x;
  int lane = tid & 31;
  int wave = tid >> 5;      // 0..7
  int wm = wave & 3;        // 16-row sub-tile
  int wn = wave >> 2;       // 0/1 -> 64-col half
  int bm = blockIdx.y * GBM;
  int bn = blockIdx.x * GBN;

  v8f acc[4];
  #pragma unroll
  for (int nt = 0; nt < 4; ++nt)
    #pragma unroll
    for (int i = 0; i < 8; ++i) acc[nt][i] = 0.0f;

  int mrow = lane & 15;
  int kb = (lane >> 4) * 8;

  // per-thread staging coordinates (branchless / in-bounds by contract)
  int arow = tid >> 2;              // 0..63
  int ac8  = (tid & 3) * 8;         // 0/8/16/24
  int bcol0 = tid >> 1;             // 0..127
  int bc8a = (tid & 1) * 8;         // halfs [bc8a, bc8a+8)
  int bc8b = bc8a + 16;             // other 16-half group of the K-step

  const _Float16* aptr  = A + (size_t)(bm + arow) * K + ac8;
  const _Float16* bptr  = Bt + (size_t)(bn + bcol0) * K + bc8a;
  const _Float16* bptr2 = Bt + (size_t)(bn + bcol0) * K + bc8b;
  int aoff = arow * LSH + ac8;
  int boff = bcol0 * LSH + bc8a;
  int boff2 = bcol0 * LSH + bc8b;

  // prologue: stage K-step 0 into buffer 0
  stage_b128(aptr, &As[0][aoff]);
  stage_b128(bptr, &Bs[0][boff]);
  stage_b128(bptr2, &Bs[0][boff2]);

  for (int k0 = 0; k0 < K; k0 += GBK) {
    int cur = (k0 >> 5) & 1;
#if USE_ASYNC_STAGE
    asm volatile("s_wait_asynccnt 0x0" ::: "memory");
#endif
    __syncthreads();   // buffer 'cur' fully staged; previous reads of 'cur^1' done

    int k1 = k0 + GBK;
    if (k1 < K) {      // stage next K-step into the other buffer (behind wmma)
      stage_b128(aptr + k1, &As[cur ^ 1][aoff]);
      stage_b128(bptr + k1, &Bs[cur ^ 1][boff]);
      stage_b128(bptr2 + k1, &Bs[cur ^ 1][boff2]);
      if (k1 + GBK < K)
        __builtin_prefetch((const void*)(aptr + k1 + GBK), 0, 1);
    }

    const _Float16* ap = &As[cur][(wm * 16 + mrow) * LSH + kb];
    v8h alo = *(const v8h*)ap;
    v8h ahi = *(const v8h*)(ap + 16);
    v16h a;
    #pragma unroll
    for (int i = 0; i < 8; ++i) { a[i] = alo[i]; a[i + 8] = ahi[i]; }

    #pragma unroll
    for (int nt = 0; nt < 4; ++nt) {
      const _Float16* bp = &Bs[cur][(wn * 64 + nt * 16 + mrow) * LSH + kb];
      v8h blo = *(const v8h*)bp;
      v8h bhi = *(const v8h*)(bp + 16);
      v16h b;
      #pragma unroll
      for (int i = 0; i < 8; ++i) { b[i] = blo[i]; b[i + 8] = bhi[i]; }
      acc[nt] = __builtin_amdgcn_wmma_f32_16x16x32_f16(
          false, a, false, b, (short)0, acc[nt], false, false);
    }
  }

  // store: lanes 0-15 -> M=r, lanes 16-31 -> M=r+8; col-in-tile = lane&15
  int rofs = (lane >> 4) * 8;
  #pragma unroll
  for (int nt = 0; nt < 4; ++nt) {
    int col = bn + wn * 64 + nt * 16 + mrow;
    if (col >= N) continue;
    float bval = bias ? bias[col] : 0.0f;
    #pragma unroll
    for (int r = 0; r < 8; ++r) {
      int row = bm + wm * 16 + r + rofs;
      float vv = acc[nt][r] + bval;
      if (flags & GF_RELU) vv = fmaxf(vv, 0.0f);
      size_t idx = (size_t)row * N + col;
      if (flags & GF_RES) vv += resid[idx];
      C[idx] = vv;
    }
  }
}

// ---------------------------------------------------------------------------
// CCA output re-alignment: h[b,s,:] += oproj[b, s-(CL-1), :] for s >= CL-1.
// ---------------------------------------------------------------------------
__global__ void k_shift_add(float* __restrict__ h, const float* __restrict__ op,
                            long long n) {
  long long t = (long long)blockIdx.x * 256 + threadIdx.x;
  if (t >= n) return;
  int d = (int)(t % DMODEL); long long r = t / DMODEL;
  int s = (int)(r % SEQ); int b = (int)(r / SEQ);
  if (s >= CHL - 1)
    h[t] += op[((long long)b * SEQ + (s - (CHL - 1))) * DMODEL + d];
}

// ---------------------------------------------------------------------------
// Host orchestration
// ---------------------------------------------------------------------------
extern "C" void kernel_launch(void* const* d_in, const int* in_sizes, int n_in,
                              void* d_out, int out_size, void* d_ws, size_t ws_size,
                              hipStream_t stream) {
  (void)in_sizes; (void)n_in; (void)out_size; (void)ws_size;
  // Input order: jax tree_leaves (dict keys sorted at every level).
  const float* E_IN   = (const float*)d_in[0];   // e [2,16,2,128,768]
  const float* H_IN   = (const float*)d_in[1];   // h [2,1024,768]
  const float* B_READ = (const float*)d_in[2];
  const float* CCA_BKV = (const float*)d_in[3];
  const float* CCA_BO  = (const float*)d_in[4];
  const float* CCA_BQ  = (const float*)d_in[5];
  const float* CCA_NRM = (const float*)d_in[6];
  const float* CCA_WKV = (const float*)d_in[7];
  const float* CCA_WO  = (const float*)d_in[8];
  const float* CCA_WQ  = (const float*)d_in[9];
  const float* DA_BO   = (const float*)d_in[10];
  const float* DA_BQKV = (const float*)d_in[11];
  const float* DA_NRM  = (const float*)d_in[12];
  const float* DA_WO   = (const float*)d_in[13];
  const float* DA_WQKV = (const float*)d_in[14];
  const float* DF_B1   = (const float*)d_in[15];
  const float* DF_B2   = (const float*)d_in[16];
  const float* DF_NRM  = (const float*)d_in[17];
  const float* DF_W1   = (const float*)d_in[18];
  const float* DF_W2   = (const float*)d_in[19];
  const float* EA_BO   = (const float*)d_in[20];
  const float* EA_BQKV = (const float*)d_in[21];
  const float* EA_NRM  = (const float*)d_in[22];
  const float* EA_WO   = (const float*)d_in[23];
  const float* EA_WQKV = (const float*)d_in[24];
  const float* ECA_BKV = (const float*)d_in[25];
  const float* ECA_BO  = (const float*)d_in[26];
  const float* ECA_BQ  = (const float*)d_in[27];
  const float* ECA_NRM = (const float*)d_in[28];
  const float* ECA_WKV = (const float*)d_in[29];
  const float* ECA_WO  = (const float*)d_in[30];
  const float* ECA_WQ  = (const float*)d_in[31];
  const float* EF_B1   = (const float*)d_in[32];
  const float* EF_B2   = (const float*)d_in[33];
  const float* EF_NRM  = (const float*)d_in[34];
  const float* EF_W1   = (const float*)d_in[35];
  const float* EF_W2   = (const float*)d_in[36];
  const float* E_NRM_H = (const float*)d_in[37];
  const float* W_READ  = (const float*)d_in[38];

  // ---- workspace bump allocator ----
  size_t off = 0;
  char* ws = (char*)d_ws;
  auto alloc_f = [&](size_t n) -> float* {
    float* p = (float*)(ws + off);
    off += ((n * 4 + 255) / 256) * 256;
    return p;
  };
  auto alloc_h = [&](size_t n) -> _Float16* {
    _Float16* p = (_Float16*)(ws + off);
    off += ((n * 2 + 255) / 256) * 256;
    return p;
  };

  const int ROWS_H = NB * SEQ;                     // 2048
  const int ROWS_E = NB * NCH * NNB * NL_;         // 8192
  const int HD = NH * DK_;                         // 768
  const int VPAD = ((VOCAB + 127) / 128) * 128;    // 50304

  float* hbuf    = alloc_f((size_t)ROWS_H * DMODEL);
  float* qkvbuf  = alloc_f((size_t)ROWS_E * 3 * HD);
  float* attnout = alloc_f((size_t)ROWS_E * HD);
  float* ffnmid  = alloc_f((size_t)ROWS_E * FFN_);
  float* ework   = alloc_f((size_t)ROWS_E * DMODEL);
  float* caq     = alloc_f((size_t)ROWS_E * HD);
  float* cakv    = alloc_f((size_t)ROWS_E * 2 * HD);
  _Float16* x16  = alloc_h((size_t)ROWS_E * FFN_);
  _Float16* hs16 = alloc_h((size_t)ROWS_H * DMODEL);

  // f16 transposed weight copies [Npad x K] per layer
  _Float16* dwqkv16 = alloc_h(4ull * 3 * HD * DMODEL);
  _Float16* dwo16   = alloc_h(4ull * DMODEL * HD);
  _Float16* dw1_16  = alloc_h(4ull * FFN_ * DMODEL);
  _Float16* dw2_16  = alloc_h(4ull * DMODEL * FFN_);
  _Float16* eaqkv16 = alloc_h(2ull * 3 * HD * DMODEL);
  _Float16* eawo16  = alloc_h(2ull * DMODEL * HD);
  _Float16* ew1_16  = alloc_h(2ull * FFN_ * DMODEL);
  _Float16* ew2_16  = alloc_h(2ull * DMODEL * FFN_);
  _Float16* ecawq16  = alloc_h((size_t)HD * DMODEL);
  _Float16* ecawkv16 = alloc_h(2ull * HD * DMODEL);
  _Float16* ecawo16  = alloc_h((size_t)DMODEL * HD);
  _Float16* ccawq16  = alloc_h((size_t)HD * DMODEL);
  _Float16* ccawkv16 = alloc_h(2ull * HD * DMODEL);
  _Float16* ccawo16  = alloc_h((size_t)DMODEL * HD);
  _Float16* wread16  = alloc_h((size_t)VPAD * DMODEL);

  // ---- launch helpers ----
  auto cvt = [&](const float* s, _Float16* d, long long n) {
    k_cvt_f16<<<dim3((unsigned)((n + 255) / 256)), dim3(256), 0, stream>>>(s, d, n);
  };
  auto cvtt = [&](const float* s, _Float16* d, int K, int N, int batch) {
    int Npad = ((N + 127) / 128) * 128;
    dim3 g((K + 31) / 32, (Npad + 31) / 32, batch);
    k_cvt_tr<<<g, dim3(256), 0, stream>>>(s, d, K, N, Npad);
  };
  auto gemm = [&](const _Float16* A, const _Float16* Bt, const float* bias,
                  const float* res, float* C, int M, int N, int K, int flags) {
    dim3 g((N + GBN - 1) / GBN, (M + GBM - 1) / GBM);
    k_gemm_f16<<<g, dim3(256), 0, stream>>>(A, Bt, bias, res, C, M, N, K, flags);
  };
  auto rms = [&](const float* x, const float* gam, _Float16* out, int rows,
                 int perBatch, int valid, int shift, int srcPerBatch) {
    k_rmsnorm<<<dim3(rows), dim3(256), 0, stream>>>(x, gam, out, perBatch, valid,
                                                    shift, srcPerBatch);
  };
  auto rope = [&](float* buf, long long rows, int seqLen) {
    long long total = rows * (long long)(NH * 32 * 2);
    k_rope<<<dim3((unsigned)((total + 255) / 256)), dim3(256), 0, stream>>>(
        buf, rows, seqLen);
  };
  auto attn = [&](const float* q, const float* k, const float* v, float* o,
                  int G, int Lq, int Lkv, int kvDiv, int causal,
                  long long qRow, long long qBatch, long long kRow, long long kBatch,
                  long long vRow, long long vBatch, long long oRow, long long oBatch) {
    k_attn<<<dim3(G, Lq), dim3(256), 0, stream>>>(
        q, k, v, o, Lq, Lkv, kvDiv, causal, 0.125f,
        qRow, qBatch, kRow, kBatch, vRow, vBatch, oRow, oBatch);
  };

  // ---- convert + transpose weights to f16 [N x K] ----
  cvtt(DA_WQKV, dwqkv16, DMODEL, 3 * HD, 4);
  cvtt(DA_WO,   dwo16,   HD, DMODEL, 4);
  cvtt(DF_W1,   dw1_16,  DMODEL, FFN_, 4);
  cvtt(DF_W2,   dw2_16,  FFN_, DMODEL, 4);
  cvtt(EA_WQKV, eaqkv16, DMODEL, 3 * HD, 2);
  cvtt(EA_WO,   eawo16,  HD, DMODEL, 2);
  cvtt(EF_W1,   ew1_16,  DMODEL, FFN_, 2);
  cvtt(EF_W2,   ew2_16,  FFN_, DMODEL, 2);
  cvtt(ECA_WQ,  ecawq16,  DMODEL, HD, 1);
  cvtt(ECA_WKV, ecawkv16, DMODEL, 2 * HD, 1);
  cvtt(ECA_WO,  ecawo16,  HD, DMODEL, 1);
  cvtt(CCA_WQ,  ccawq16,  DMODEL, HD, 1);
  cvtt(CCA_WKV, ccawkv16, DMODEL, 2 * HD, 1);
  cvtt(CCA_WO,  ccawo16,  HD, DMODEL, 1);
  cvtt(W_READ,  wread16,  DMODEL, VOCAB, 1);

  hipMemcpyAsync(hbuf, H_IN, sizeof(float) * (size_t)ROWS_H * DMODEL,
                 hipMemcpyDeviceToDevice, stream);

  // ---- decoder layers ----
  for (int i = 0; i < 4; ++i) {
    // self attention (causal)
    rms(hbuf, DA_NRM + i * DMODEL, x16, ROWS_H, ROWS_H, ROWS_H, 0, ROWS_H);
    gemm(x16, dwqkv16 + (size_t)i * 3 * HD * DMODEL, DA_BQKV + i * 3 * HD,
         nullptr, qkvbuf, ROWS_H, 3 * HD, DMODEL, 0);
    rope(qkvbuf, ROWS_H, SEQ);
    attn(qkvbuf, qkvbuf + HD, qkvbuf + 2 * HD, attnout,
         NB * NH, SEQ, SEQ, 1, 1,
         3 * HD, (long long)SEQ * 3 * HD, 3 * HD, (long long)SEQ * 3 * HD,
         3 * HD, (long long)SEQ * 3 * HD, HD, (long long)SEQ * HD);
    cvt(attnout, x16, (long long)ROWS_H * HD);
    gemm(x16, dwo16 + (size_t)i * DMODEL * HD, DA_BO + i * DMODEL,
         hbuf, hbuf, ROWS_H, DMODEL, HD, GF_RES);

    if (i == 2) {
      // ---- retrieval encoder ----
      hipMemcpyAsync(ework, E_IN, sizeof(float) * (size_t)ROWS_E * DMODEL,
                     hipMemcpyDeviceToDevice, stream);
      rms(hbuf, E_NRM_H, hs16, ROWS_H, ROWS_H, ROWS_H, 0, ROWS_H);  // hs
      for (int ei = 0; ei < 2; ++ei) {
        // encoder self attention over (B*C*N, L, D), non-causal
        rms(ework, EA_NRM + ei * DMODEL, x16, ROWS_E, ROWS_E, ROWS_E, 0, ROWS_E);
        gemm(x16, eaqkv16 + (size_t)ei * 3 * HD * DMODEL, EA_BQKV + ei * 3 * HD,
             nullptr, qkvbuf, ROWS_E, 3 * HD, DMODEL, 0);
        rope(qkvbuf, ROWS_E, NL_);
        attn(qkvbuf, qkvbuf + HD, qkvbuf + 2 * HD, attnout,
             NB * NCH * NNB * NH, NL_, NL_, 1, 0,
             3 * HD, (long long)NL_ * 3 * HD, 3 * HD, (long long)NL_ * 3 * HD,
             3 * HD, (long long)NL_ * 3 * HD, HD, (long long)NL_ * HD);
        cvt(attnout, x16, (long long)ROWS_E * HD);
        gemm(x16, eawo16 + (size_t)ei * DMODEL * HD, EA_BO + ei * DMODEL,
             ework, ework, ROWS_E, DMODEL, HD, GF_RES);

        if (ei == 1) {
          // encoder cross attention: q from e, kv from hs
          rms(ework, ECA_NRM, x16, ROWS_E, ROWS_E, ROWS_E, 0, ROWS_E);
          gemm(x16, ecawq16, ECA_BQ, nullptr, caq, ROWS_E, HD, DMODEL, 0);
          gemm(hs16, ecawkv16, ECA_BKV, nullptr, cakv, ROWS_H, 2 * HD, DMODEL, 0);
          attn(caq, cakv, cakv + HD, attnout,
               NB * NCH * NNB * NH, NL_, CHL, NNB, 0,
               HD, (long long)NL_ * HD, 2 * HD, (long long)CHL * 2 * HD,
               2 * HD, (long long)CHL * 2 * HD, HD, (long long)NL_ * HD);
          cvt(attnout, x16, (long long)ROWS_E * HD);
          gemm(x16, ecawo16, ECA_BO, ework, ework, ROWS_E, DMODEL, HD, GF_RES);
        }
        // encoder FFN
        rms(ework, EF_NRM + ei * DMODEL, x16, ROWS_E, ROWS_E, ROWS_E, 0, ROWS_E);
        gemm(x16, ew1_16 + (size_t)ei * FFN_ * DMODEL, EF_B1 + ei * FFN_,
             nullptr, ffnmid, ROWS_E, FFN_, DMODEL, GF_RELU);
        cvt(ffnmid, x16, (long long)ROWS_E * FFN_);
        gemm(x16, ew2_16 + (size_t)ei * DMODEL * FFN_, EF_B2 + ei * DMODEL,
             ework, ework, ROWS_E, DMODEL, FFN_, GF_RES);
      }

      // ---- chunked cross attention ----
      // x = rmsnorm(h[:, CL-1:]) padded to [B, C*CL, D]
      rms(hbuf, CCA_NRM, x16, ROWS_H, SEQ, SEQ - (CHL - 1), CHL - 1, SEQ);
      gemm(x16, ccawq16, CCA_BQ, nullptr, caq, ROWS_H, HD, DMODEL, 0);
      cvt(ework, x16, (long long)ROWS_E * DMODEL);   // e_enc -> f16
      gemm(x16, ccawkv16, CCA_BKV, nullptr, cakv, ROWS_E, 2 * HD, DMODEL, 0);
      attn(caq, cakv, cakv + HD, attnout,
           NB * NCH * NH, CHL, NNB * NL_, 1, 0,
           HD, (long long)CHL * HD, 2 * HD, (long long)NNB * NL_ * 2 * HD,
           2 * HD, (long long)NNB * NL_ * 2 * HD, HD, (long long)CHL * HD);
      cvt(attnout, x16, (long long)ROWS_H * HD);
      gemm(x16, ccawo16, CCA_BO, nullptr, ffnmid, ROWS_H, DMODEL, HD, 0);
      {
        long long n = (long long)ROWS_H * DMODEL;
        k_shift_add<<<dim3((unsigned)((n + 255) / 256)), dim3(256), 0, stream>>>(
            hbuf, ffnmid, n);
      }
    }

    // decoder FFN
    rms(hbuf, DF_NRM + i * DMODEL, x16, ROWS_H, ROWS_H, ROWS_H, 0, ROWS_H);
    gemm(x16, dw1_16 + (size_t)i * FFN_ * DMODEL, DF_B1 + i * FFN_,
         nullptr, ffnmid, ROWS_H, FFN_, DMODEL, GF_RELU);
    cvt(ffnmid, x16, (long long)ROWS_H * FFN_);
    gemm(x16, dw2_16 + (size_t)i * DMODEL * FFN_, DF_B2 + i * DMODEL,
         hbuf, hbuf, ROWS_H, DMODEL, FFN_, GF_RES);
  }

  // ---- readout: logits = h @ w_read + b_read ----
  cvt(hbuf, x16, (long long)ROWS_H * DMODEL);
  gemm(x16, wread16, B_READ, nullptr, (float*)d_out, ROWS_H, VOCAB, DMODEL, 0);
}
